// SAGEFC_326417514546
// MI455X (gfx1250) — matrix operand order repaired and checked by reading, not compiled, not run on hardware
//
#include <hip/hip_runtime.h>

// ---------------------------------------------------------------------------
// Types for CDNA5 WMMA (wave32, bf16 -> f32 accumulate)
// ---------------------------------------------------------------------------
typedef __bf16 bf16;
typedef __attribute__((ext_vector_type(16))) __bf16 v16bf;
typedef __attribute__((ext_vector_type(8)))  __bf16 v8bf;
typedef __attribute__((ext_vector_type(8)))  float  v8f;

#define N_NODES 100000
#define N_PAD   100096   // multiple of 128
#define N_EDGES 1600000
#define EPS     1e-5f

// ---------------------------------------------------------------------------
// Utility kernels
// ---------------------------------------------------------------------------
__global__ void fill_f32(float* __restrict__ p, float v, size_t n) {
    size_t i = (size_t)blockIdx.x * blockDim.x + threadIdx.x;
    size_t stride = (size_t)gridDim.x * blockDim.x;
    for (; i < n; i += stride) p[i] = v;
}

// dst is [rows_pad, cols_dst] bf16; src is [rows, cols_src] f32; zero-pads.
__global__ void cast_pad(bf16* __restrict__ dst, const float* __restrict__ src,
                         int rows, int cols_src, int cols_dst, int rows_pad) {
    long i = (long)blockIdx.x * blockDim.x + threadIdx.x;
    long total = (long)rows_pad * cols_dst;
    if (i >= total) return;
    int r = (int)(i / cols_dst);
    int c = (int)(i - (long)r * cols_dst);
    float v = (r < rows && c < cols_src) ? src[(size_t)r * cols_src + c] : 0.0f;
    dst[i] = (bf16)v;
}

__global__ void deg_count(const int* __restrict__ edst, float* __restrict__ deg, int E) {
    int e = blockIdx.x * blockDim.x + threadIdx.x;
    if (e < E) atomicAdd(deg + edst[e], 1.0f);
}

__global__ void deg_inv_k(float* __restrict__ deg, int n) {
    int i = blockIdx.x * blockDim.x + threadIdx.x;
    if (i < n) deg[i] = 1.0f / fmaxf(deg[i], 1.0f);
}

// Edge scatter: each thread moves 8 bf16 features of one edge (one wave = one
// 256-wide edge row). bf16 gather halves HBM read traffic; accumulate in f32.
__global__ void scatter_add(const int* __restrict__ esrc, const int* __restrict__ edst,
                            const bf16* __restrict__ xb, float* __restrict__ agg,
                            int E, int D) {
    long tid = (long)blockIdx.x * blockDim.x + threadIdx.x;
    int per = D >> 3;                      // 8 elems per thread
    long e = tid / per;
    if (e >= E) return;
    int off = (int)(tid - e * per) << 3;
    int s = esrc[e];
    int d = edst[e];
    v8bf v = *(const v8bf*)(xb + (size_t)s * D + off);
    float* a = agg + (size_t)d * D + off;
#pragma unroll
    for (int i = 0; i < 8; ++i) atomicAdd(a + i, (float)v[i]);
}

// aggb = bf16(agg * (1/deg[row]))
__global__ void scale_cast(const float* __restrict__ agg, const float* __restrict__ dinv,
                           bf16* __restrict__ aggb, int rows, int D) {
    long i = (long)blockIdx.x * blockDim.x + threadIdx.x;
    if (i >= (long)rows * D) return;
    int r = (int)(i / D);
    aggb[i] = (bf16)(agg[i] * dinv[r]);
}

// ---------------------------------------------------------------------------
// WMMA fragment loaders (layouts per CDNA5 ISA 7.12.2, wave32)
// ---------------------------------------------------------------------------
// A (16x32 bf16, row-major source, m already includes lane&15):
// lanes 0-15 hold K = k0+0..7 and k0+16..23 ; lanes 16-31 hold K = k0+8..15, k0+24..31
__device__ __forceinline__ v16bf load_a(const bf16* __restrict__ base, int ld,
                                        int m, int k0, int lane) {
    const bf16* p = base + (size_t)m * ld + k0 + ((lane >> 4) << 3);
    v8bf lo = *(const v8bf*)p;
    v8bf hi = *(const v8bf*)(p + 16);
    v16bf r;
#pragma unroll
    for (int i = 0; i < 8; ++i) { r[i] = lo[i]; r[8 + i] = hi[i]; }
    return r;
}

// B (32x16 bf16) built from row-major weight W[DOUT, DIN]: B[k][n] = W[n*ld + k].
// lane<16: column n0+lane, K = k0+0..7, k0+16..23; lane>=16: K = k0+8..15, k0+24..31.
__device__ __forceinline__ v16bf load_b(const bf16* __restrict__ W, int ld,
                                        int n0, int k0, int lane) {
    const bf16* p = W + (size_t)(n0 + (lane & 15)) * ld + k0 + ((lane >> 4) << 3);
    v8bf lo = *(const v8bf*)p;
    v8bf hi = *(const v8bf*)(p + 16);
    v16bf r;
#pragma unroll
    for (int i = 0; i < 8; ++i) { r[i] = lo[i]; r[8 + i] = hi[i]; }
    return r;
}

// ---------------------------------------------------------------------------
// Fused dual GEMM:  out = [relu]( Aagg @ Wl^T + Ax @ Wr^T + bias )
// Block = 256 threads = 8 waves; each wave owns a 16(M) x 64(N) strip.
// ---------------------------------------------------------------------------
template <int DIN, int DOUT, bool DUAL, bool RELU>
__global__ void gemm_wmma(const bf16* __restrict__ Aagg, const bf16* __restrict__ Ax,
                          const bf16* __restrict__ Wl, const bf16* __restrict__ Wr,
                          const float* __restrict__ bias,
                          bf16* __restrict__ outb, float* __restrict__ outf, int nreal) {
    const int lane = threadIdx.x & 31;
    const int wave = threadIdx.x >> 5;
    const int m0 = blockIdx.x * 128 + wave * 16;
    const int n0 = blockIdx.y * 64;
    const int m = m0 + (lane & 15);

    v8f acc[4] = {};

    for (int k0 = 0; k0 < DIN; k0 += 32) {
        v16bf a0 = load_a(Aagg, DIN, m, k0, lane);
        v16bf a1;
        if (DUAL) a1 = load_a(Ax, DIN, m, k0, lane);
#pragma unroll
        for (int t = 0; t < 4; ++t) {
            v16bf bwl = load_b(Wl, DIN, n0 + 16 * t, k0, lane);
            acc[t] = __builtin_amdgcn_wmma_f32_16x16x32_bf16(
                false, a0, false, bwl, (short)0, acc[t], false, false);
            if (DUAL) {
                v16bf bwr = load_b(Wr, DIN, n0 + 16 * t, k0, lane);
                acc[t] = __builtin_amdgcn_wmma_f32_16x16x32_bf16(
                    false, a1, false, bwr, (short)0, acc[t], false, false);
            }
        }
    }

    // C/D layout: lane<16 -> rows m0..m0+7, lane>=16 -> rows m0+8..m0+15; col = n0t + lane&15
    const int mb = m0 + ((lane >> 4) << 3);
#pragma unroll
    for (int t = 0; t < 4; ++t) {
        int n = n0 + 16 * t + (lane & 15);
        float bv = bias[n];
#pragma unroll
        for (int r = 0; r < 8; ++r) {
            float v = acc[t][r] + bv;
            if (RELU) v = fmaxf(v, 0.0f);
            int row = mb + r;
            if (outb) outb[(size_t)row * DOUT + n] = (bf16)v;
            if (outf && row < nreal) outf[(size_t)row * DOUT + n] = v;
        }
    }
}

// ---------------------------------------------------------------------------
// BatchNorm: per-block partial column sums, then normalize
// ---------------------------------------------------------------------------
__global__ void bn_stats(const float* __restrict__ z, float* __restrict__ stats, int nrows) {
    int c = threadIdx.x;                     // 128 threads = 128 channels
    int r0 = blockIdx.x * 256;
    int rend = r0 + 256; if (rend > nrows) rend = nrows;
    float s = 0.0f, s2 = 0.0f;
    for (int r = r0; r < rend; ++r) {
        float v = z[(size_t)r * 128 + c];
        s += v; s2 += v * v;
    }
    atomicAdd(stats + c, s);
    atomicAdd(stats + 128 + c, s2);
}

__global__ void bn_apply(const float* __restrict__ z, const float* __restrict__ stats,
                         const float* __restrict__ gamma, const float* __restrict__ beta,
                         float* __restrict__ out, int nrows) {
    long i = (long)blockIdx.x * blockDim.x + threadIdx.x;
    if (i >= (long)nrows * 128) return;
    int c = (int)(i & 127);
    float inv_n = 1.0f / (float)nrows;
    float mu = stats[c] * inv_n;
    float var = stats[128 + c] * inv_n - mu * mu;
    out[i] = gamma[c] * (z[i] - mu) * rsqrtf(var + EPS) + beta[c];
}

// ---------------------------------------------------------------------------
// Host-side orchestration
// ---------------------------------------------------------------------------
extern "C" void kernel_launch(void* const* d_in, const int* in_sizes, int n_in,
                              void* d_out, int out_size, void* d_ws, size_t ws_size,
                              hipStream_t stream) {
    const float* x    = (const float*)d_in[0];
    const int*   ei   = (const int*)d_in[1];
    const float* Wl0  = (const float*)d_in[2];
    const float* bl0  = (const float*)d_in[3];
    const float* Wr0  = (const float*)d_in[4];
    const float* Wl1  = (const float*)d_in[5];
    const float* bl1  = (const float*)d_in[6];
    const float* Wr1  = (const float*)d_in[7];
    const float* Wl2  = (const float*)d_in[8];
    const float* bl2  = (const float*)d_in[9];
    const float* Wr2  = (const float*)d_in[10];
    const float* Wp   = (const float*)d_in[11];
    const float* bp   = (const float*)d_in[12];
    const float* gamma= (const float*)d_in[13];
    const float* beta = (const float*)d_in[14];

    const int* esrc = ei;            // edge_index[0]
    const int* edst = ei + N_EDGES;  // edge_index[1]

    // carve workspace
    char* ws = (char*)d_ws;
    size_t off = 0;
    auto alloc = [&](size_t bytes) -> void* {
        void* p = ws + off;
        off += (bytes + 255) & ~(size_t)255;
        return p;
    };
    bf16* actA = (bf16*)alloc((size_t)N_PAD * 256 * sizeof(bf16));
    bf16* actB = (bf16*)alloc((size_t)N_PAD * 256 * sizeof(bf16));
    float* aggf = (float*)alloc((size_t)N_PAD * 256 * sizeof(float));
    bf16* aggb = (bf16*)alloc((size_t)N_PAD * 256 * sizeof(bf16));
    float* zbuf = (float*)alloc((size_t)N_PAD * 128 * sizeof(float));
    float* dinv = (float*)alloc((size_t)N_PAD * sizeof(float));
    float* stats = (float*)alloc(256 * sizeof(float));
    bf16* wl0b = (bf16*)alloc((size_t)256 * 128 * sizeof(bf16));
    bf16* wr0b = (bf16*)alloc((size_t)256 * 128 * sizeof(bf16));
    bf16* wl1b = (bf16*)alloc((size_t)256 * 256 * sizeof(bf16));
    bf16* wr1b = (bf16*)alloc((size_t)256 * 256 * sizeof(bf16));
    bf16* wl2b = (bf16*)alloc((size_t)256 * 256 * sizeof(bf16));
    bf16* wr2b = (bf16*)alloc((size_t)256 * 256 * sizeof(bf16));
    bf16* wpb  = (bf16*)alloc((size_t)128 * 256 * sizeof(bf16));

    const int TB = 256;
    auto blocks = [](long n, int tb) { return (unsigned)((n + tb - 1) / tb); };

    // weights -> bf16 (layer 0 padded K: 100 -> 128)
    cast_pad<<<blocks((long)256 * 128, TB), TB, 0, stream>>>(wl0b, Wl0, 256, 100, 128, 256);
    cast_pad<<<blocks((long)256 * 128, TB), TB, 0, stream>>>(wr0b, Wr0, 256, 100, 128, 256);
    cast_pad<<<blocks((long)256 * 256, TB), TB, 0, stream>>>(wl1b, Wl1, 256, 256, 256, 256);
    cast_pad<<<blocks((long)256 * 256, TB), TB, 0, stream>>>(wr1b, Wr1, 256, 256, 256, 256);
    cast_pad<<<blocks((long)256 * 256, TB), TB, 0, stream>>>(wl2b, Wl2, 256, 256, 256, 256);
    cast_pad<<<blocks((long)256 * 256, TB), TB, 0, stream>>>(wr2b, Wr2, 256, 256, 256, 256);
    cast_pad<<<blocks((long)128 * 256, TB), TB, 0, stream>>>(wpb, Wp, 128, 256, 256, 128);

    // degrees -> 1/max(deg,1)
    fill_f32<<<1024, TB, 0, stream>>>(dinv, 0.0f, (size_t)N_PAD);
    deg_count<<<blocks(N_EDGES, TB), TB, 0, stream>>>(edst, dinv, N_EDGES);
    deg_inv_k<<<blocks(N_PAD, TB), TB, 0, stream>>>(dinv, N_PAD);

    // x (N,100) f32 -> actA (N_PAD,128) bf16, zero padded
    cast_pad<<<blocks((long)N_PAD * 128, TB), TB, 0, stream>>>(actA, x, N_NODES, 100, 128, N_PAD);

    dim3 g256(N_PAD / 128, 4);  // DOUT=256 -> 4 column tiles of 64
    dim3 g128(N_PAD / 128, 2);  // DOUT=128

    // ---- layer 0: Din=128(pad), Dout=256, relu ----
    fill_f32<<<4096, TB, 0, stream>>>(aggf, 0.0f, (size_t)N_PAD * 128);
    scatter_add<<<blocks((long)N_EDGES * (128 / 8), TB), TB, 0, stream>>>(esrc, edst, actA, aggf, N_EDGES, 128);
    scale_cast<<<blocks((long)N_PAD * 128, TB), TB, 0, stream>>>(aggf, dinv, aggb, N_PAD, 128);
    gemm_wmma<128, 256, true, true><<<g256, TB, 0, stream>>>(aggb, actA, wl0b, wr0b, bl0, actB, nullptr, N_PAD);

    // ---- layer 1: Din=256, Dout=256, relu ----
    fill_f32<<<4096, TB, 0, stream>>>(aggf, 0.0f, (size_t)N_PAD * 256);
    scatter_add<<<blocks((long)N_EDGES * (256 / 8), TB), TB, 0, stream>>>(esrc, edst, actB, aggf, N_EDGES, 256);
    scale_cast<<<blocks((long)N_PAD * 256, TB), TB, 0, stream>>>(aggf, dinv, aggb, N_PAD, 256);
    gemm_wmma<256, 256, true, true><<<g256, TB, 0, stream>>>(aggb, actB, wl1b, wr1b, bl1, actA, nullptr, N_PAD);

    // ---- layer 2: Din=256, Dout=256, no relu; f32 rows -> d_out[0 : N*256) ----
    fill_f32<<<4096, TB, 0, stream>>>(aggf, 0.0f, (size_t)N_PAD * 256);
    scatter_add<<<blocks((long)N_EDGES * (256 / 8), TB), TB, 0, stream>>>(esrc, edst, actA, aggf, N_EDGES, 256);
    scale_cast<<<blocks((long)N_PAD * 256, TB), TB, 0, stream>>>(aggf, dinv, aggb, N_PAD, 256);
    gemm_wmma<256, 256, true, false><<<g256, TB, 0, stream>>>(aggb, actA, wl2b, wr2b, bl2, actB,
                                                              (float*)d_out, N_NODES);

    // ---- projection head: z = x2 @ Wp^T + bp  (single-A GEMM, f32 out) ----
    gemm_wmma<256, 128, false, false><<<g128, TB, 0, stream>>>(actB, nullptr, wpb, nullptr, bp,
                                                               nullptr, zbuf, N_PAD);

    // ---- BatchNorm over channels -> d_out[N*256 : N*256 + N*128) ----
    fill_f32<<<1, 256, 0, stream>>>(stats, 0.0f, 256);
    bn_stats<<<blocks(N_NODES, 256), 128, 0, stream>>>(zbuf, stats, N_NODES);
    bn_apply<<<blocks((long)N_NODES * 128, TB), TB, 0, stream>>>(
        zbuf, stats, gamma, beta, (float*)d_out + (size_t)N_NODES * 256, N_NODES);
}
